// CMTBlock_73967926771818
// MI455X (gfx1250) — matrix operand order repaired and checked by reading, not compiled, not run on hardware
//
#include <hip/hip_runtime.h>
#include <hip/hip_bf16.h>
#include <math.h>

// ---------------- CMT block for MI455X (gfx1250, wave32, WMMA) ----------------
// B=32, C=256, H=W=28 (P=784), HEADS=8, DK=DV=32, Nk=196 (padded to 224), R=4 (Cm=1024)

#define BB    32
#define CC    256
#define PP    784      // 28*28
#define WW    28
#define CHW   200704   // 256*784
#define NH    8
#define NKV   196
#define NKP   224      // 14*16, key padding
#define CM    1024
#define EPSL  1e-5f
#define SCALE 0.17677669529663687f   // 32^-0.5

typedef __attribute__((ext_vector_type(16))) _Float16 v16h;
typedef __attribute__((ext_vector_type(8)))  _Float16 v8h;
typedef __attribute__((ext_vector_type(8)))  float    v8f;

// ---- WMMA fragment loaders (contiguous k order, same in A and B) ----
// A: 16x32 f16 (MxK), row-major, stride lda halves.
// lane<16 : M=lane,   elems0-15 = K 0..15
// lane>=16: M=lane-16,elems0-15 = K 16..31
static __device__ inline v16h load_A16x32(const _Float16* base, int lda, int lane) {
  int m  = lane & 15;
  int kb = (lane < 16) ? 0 : 16;
  const _Float16* p = base + (long)m * lda + kb;
  v8h lo = *(const v8h*)(p);
  v8h hi = *(const v8h*)(p + 8);
  v16h a;
#pragma unroll
  for (int i = 0; i < 8; ++i) { a[i] = lo[i]; a[i + 8] = hi[i]; }
  return a;
}

// B: 32x16 f16 (KxN). Memory holds rows indexed by n: base + n*ldb + k  (B(k,n)=base[n][k]).
// lane<16: K 0..15 for col N=lane ; lane>=16: K 16..31 for col N=lane-16.
static __device__ inline v16h load_B32x16(const _Float16* base, int ldb, int lane) {
  int n  = lane & 15;
  int kb = (lane < 16) ? 0 : 16;
  return *(const v16h*)(base + (long)n * ldb + kb);
}

static __device__ inline v8f wmma_f16(v16h a, v16h b, v8f c) {
  return __builtin_amdgcn_wmma_f32_16x16x32_f16(false, a, false, b, (short)0, c, false, false);
}

static __device__ inline float gelu_exact(float z) {
  return 0.5f * z * (1.0f + erff(z * 0.70710678118654752f));
}

// ======================= elementwise / prep kernels =======================

// depthwise 3x3 pad1 + bias + residual : x -> x1   (grid over B*C*P)
__global__ __launch_bounds__(256) void k_lpu(const float* __restrict__ x,
                                             const float* __restrict__ w,
                                             const float* __restrict__ bias,
                                             float* __restrict__ x1) {
  int i = blockIdx.x * 256 + threadIdx.x;
  if (i >= BB * CC * PP) return;
  int p = i % PP, bc = i / PP, c = bc % CC;
  int hh = p / WW, ww = p % WW;
  float acc = bias[c];
#pragma unroll
  for (int ky = 0; ky < 3; ++ky) {
    int y = hh + ky - 1;
    if ((unsigned)y >= WW) continue;
#pragma unroll
    for (int kx = 0; kx < 3; ++kx) {
      int xx = ww + kx - 1;
      if ((unsigned)xx >= WW) continue;
      acc += x[(long)bc * PP + y * WW + xx] * w[c * 9 + ky * 3 + kx];
    }
  }
  x1[i] = acc + x[i];
}

// per-sample layernorm stats over C*H*W  (grid=B, block=256)
__global__ __launch_bounds__(256) void k_ln_stats(const float* __restrict__ src,
                                                  float* __restrict__ stats) {
  __shared__ float ssum[256], ssq[256];
  int b = blockIdx.x;
  const float* p = src + (long)b * CHW;
  float s = 0.f, q = 0.f;
  for (int i = threadIdx.x; i < CHW; i += 256) { float v = p[i]; s += v; q += v * v; }
  ssum[threadIdx.x] = s; ssq[threadIdx.x] = q;
  __syncthreads();
  for (int st = 128; st > 0; st >>= 1) {
    if (threadIdx.x < st) { ssum[threadIdx.x] += ssum[threadIdx.x + st]; ssq[threadIdx.x] += ssq[threadIdx.x + st]; }
    __syncthreads();
  }
  if (threadIdx.x == 0) {
    float m = ssum[0] / (float)CHW;
    float var = ssq[0] / (float)CHW - m * m;
    stats[b * 2] = m;
    stats[b * 2 + 1] = rsqrtf(var + EPSL);
  }
}

// normalize CHW f32 -> token-major fp16 [b, p, c]
__global__ __launch_bounds__(256) void k_ln_pack(const float* __restrict__ src,
                                                 const float* __restrict__ stats,
                                                 _Float16* __restrict__ dst) {
  long i = (long)blockIdx.x * 256 + threadIdx.x;
  if (i >= (long)BB * CHW) return;
  int b = (int)(i / CHW), f = (int)(i % CHW);
  int c = f / PP, p = f % PP;
  float v = (src[i] - stats[b * 2]) * stats[b * 2 + 1];
  dst[((long)b * PP + p) * CC + c] = (_Float16)v;
}

// depthwise 2x2 stride2 on x1 -> kvr fp16 [b, key(0..223), c], zero-padded keys
__global__ __launch_bounds__(256) void k_kvconv(const float* __restrict__ x1,
                                                const float* __restrict__ w,
                                                const float* __restrict__ bias,
                                                _Float16* __restrict__ kvr) {
  int i = blockIdx.x * 256 + threadIdx.x;
  if (i >= BB * NKP * CC) return;
  int c = i % CC, row = i / CC;
  int key = row % NKP, b = row / NKP;
  float acc = 0.f;
  if (key < NKV) {
    int oy = key / 14, ox = key % 14;
    acc = bias[c];
#pragma unroll
    for (int ky = 0; ky < 2; ++ky)
#pragma unroll
      for (int kx = 0; kx < 2; ++kx)
        acc += x1[((long)b * CC + c) * PP + (2 * oy + ky) * WW + 2 * ox + kx] * w[c * 4 + ky * 2 + kx];
  }
  kvr[(long)row * CC + c] = (_Float16)acc;
}

// f32 -> f16 weight conversion
__global__ __launch_bounds__(256) void k_cvt16(const float* __restrict__ s,
                                               _Float16* __restrict__ d, int n) {
  int i = blockIdx.x * 256 + threadIdx.x;
  if (i < n) d[i] = (_Float16)s[i];
}

// depthwise 3x3 pad1 on t1 fp16 [b,cm,p] + BN2 + GELU -> t2 fp16 [b, p, cm]
__global__ __launch_bounds__(256) void k_dw2(const _Float16* __restrict__ t1,
                                             const float* __restrict__ w,
                                             const float* __restrict__ bias,
                                             const float* __restrict__ g2, const float* __restrict__ b2,
                                             const float* __restrict__ m2, const float* __restrict__ v2,
                                             _Float16* __restrict__ t2) {
  long i = (long)blockIdx.x * 256 + threadIdx.x;
  if (i >= (long)BB * CM * PP) return;
  int p = (int)(i % PP); long bc = i / PP;
  int cm = (int)(bc % CM); int b = (int)(bc / CM);
  int hh = p / WW, ww = p % WW;
  float acc = bias[cm];
#pragma unroll
  for (int ky = 0; ky < 3; ++ky) {
    int y = hh + ky - 1;
    if ((unsigned)y >= WW) continue;
#pragma unroll
    for (int kx = 0; kx < 3; ++kx) {
      int xx = ww + kx - 1;
      if ((unsigned)xx >= WW) continue;
      acc += (float)t1[bc * PP + y * WW + xx] * w[cm * 9 + ky * 3 + kx];
    }
  }
  float z = (acc - m2[cm]) * (g2[cm] * rsqrtf(v2[cm] + EPSL)) + b2[cm];
  z = gelu_exact(z);
  t2[((long)b * PP + p) * CM + cm] = (_Float16)z;
}

// ======================= WMMA GEMM kernels =======================
// All: one wave per 16x16 D tile; A token-major [M,K] fp16; W row-major [NCOL,K] fp16 (B(k,n)=W[n][k]).

// Q/K/V projection: dst fp16; store_mode 0 -> [bh, tok, 32] ; 1 -> [bh, d, rows_pb] (transposed, for V)
__global__ __launch_bounds__(256) void k_gemm_proj(const _Float16* __restrict__ A,
                                                   const _Float16* __restrict__ W,
                                                   const float* __restrict__ bias,
                                                   _Float16* __restrict__ dst,
                                                   int M, int K, int rows_pb, int store_mode) {
  int lane = threadIdx.x & 31, wave = threadIdx.x >> 5;
  const int ntn = CC >> 4;
  long tile = (long)blockIdx.x * 8 + wave;
  if (tile >= (long)(M >> 4) * ntn) return;
  int tn = (int)(tile % ntn), tm = (int)(tile / ntn);
  int row0 = tm << 4, col0 = tn << 4;
  v8f acc = {};
  for (int kk = 0; kk < K; kk += 32) {
    v16h a = load_A16x32(A + (long)row0 * K + kk, K, lane);
    v16h b = load_B32x16(W + (long)col0 * K + kk, K, lane);
    acc = wmma_f16(a, b, acc);
  }
  int n = lane & 15, rbase = (lane >> 4) << 3;
  int col = col0 + n;
  float bv = bias[col];
  int h = col >> 5, d = col & 31;
#pragma unroll
  for (int r = 0; r < 8; ++r) {
    int row = row0 + rbase + r;
    int b_ = row / rows_pb, tok = row % rows_pb;
    float v = acc[r] + bv;
    long idx = (store_mode == 0)
                 ? (((long)(b_ * NH + h) * rows_pb) + tok) * 32 + d
                 : ((long)(b_ * NH + h) * 32 + d) * rows_pb + tok;
    dst[idx] = (_Float16)v;
  }
}

// output projection: x2 = A @ wo^T + bo + x1   (f32, raw CHW-reinterpret layout)
__global__ __launch_bounds__(256) void k_gemm_oproj(const _Float16* __restrict__ A,
                                                    const _Float16* __restrict__ W,
                                                    const float* __restrict__ bo,
                                                    const float* __restrict__ x1,
                                                    float* __restrict__ x2, int M) {
  int lane = threadIdx.x & 31, wave = threadIdx.x >> 5;
  const int ntn = CC >> 4;
  long tile = (long)blockIdx.x * 8 + wave;
  if (tile >= (long)(M >> 4) * ntn) return;
  int tn = (int)(tile % ntn), tm = (int)(tile / ntn);
  int row0 = tm << 4, col0 = tn << 4;
  v8f acc = {};
  for (int kk = 0; kk < CC; kk += 32) {
    v16h a = load_A16x32(A + (long)row0 * CC + kk, CC, lane);
    v16h b = load_B32x16(W + (long)col0 * CC + kk, CC, lane);
    acc = wmma_f16(a, b, acc);
  }
  int n = lane & 15, rbase = (lane >> 4) << 3;
  int col = col0 + n;
  float bv = bo[col];
#pragma unroll
  for (int r = 0; r < 8; ++r) {
    long idx = (long)(row0 + rbase + r) * CC + col;  // == b*CHW + n*256 + c  (raw .view)
    x2[idx] = acc[r] + bv + x1[idx];
  }
}

// IRFFN conv1: t1 = GELU(BN1(A @ c1^T + c1_b))  -> fp16 [b, co, p]
__global__ __launch_bounds__(256) void k_gemm_ffn1(const _Float16* __restrict__ A,
                                                   const _Float16* __restrict__ W,
                                                   const float* __restrict__ cb,
                                                   const float* __restrict__ g1, const float* __restrict__ b1,
                                                   const float* __restrict__ m1, const float* __restrict__ v1,
                                                   _Float16* __restrict__ t1, int M) {
  int lane = threadIdx.x & 31, wave = threadIdx.x >> 5;
  const int ntn = CM >> 4;
  long tile = (long)blockIdx.x * 8 + wave;
  if (tile >= (long)(M >> 4) * ntn) return;
  int tn = (int)(tile % ntn), tm = (int)(tile / ntn);
  int row0 = tm << 4, col0 = tn << 4;
  v8f acc = {};
  for (int kk = 0; kk < CC; kk += 32) {
    v16h a = load_A16x32(A + (long)row0 * CC + kk, CC, lane);
    v16h b = load_B32x16(W + (long)col0 * CC + kk, CC, lane);
    acc = wmma_f16(a, b, acc);
  }
  int n = lane & 15, rbase = (lane >> 4) << 3;
  int col = col0 + n;
  float sc = g1[col] * rsqrtf(v1[col] + EPSL);
  float sh = b1[col] - m1[col] * sc;
  float bv = cb[col];
#pragma unroll
  for (int r = 0; r < 8; ++r) {
    int row = row0 + rbase + r;
    int b_ = row / PP, p = row % PP;
    float z = gelu_exact((acc[r] + bv) * sc + sh);
    t1[((long)b_ * CM + col) * PP + p] = (_Float16)z;
  }
}

// IRFFN conv2: out = x2 + BN3(A @ c2^T + c2_b)  (f32, CHW)
__global__ __launch_bounds__(256) void k_gemm_ffn2(const _Float16* __restrict__ A,
                                                   const _Float16* __restrict__ W,
                                                   const float* __restrict__ cb,
                                                   const float* __restrict__ g3, const float* __restrict__ b3,
                                                   const float* __restrict__ m3, const float* __restrict__ v3,
                                                   const float* __restrict__ x2,
                                                   float* __restrict__ out, int M) {
  int lane = threadIdx.x & 31, wave = threadIdx.x >> 5;
  const int ntn = CC >> 4;
  long tile = (long)blockIdx.x * 8 + wave;
  if (tile >= (long)(M >> 4) * ntn) return;
  int tn = (int)(tile % ntn), tm = (int)(tile / ntn);
  int row0 = tm << 4, col0 = tn << 4;
  v8f acc = {};
  for (int kk = 0; kk < CM; kk += 32) {
    v16h a = load_A16x32(A + (long)row0 * CM + kk, CM, lane);
    v16h b = load_B32x16(W + (long)col0 * CM + kk, CM, lane);
    acc = wmma_f16(a, b, acc);
  }
  int n = lane & 15, rbase = (lane >> 4) << 3;
  int col = col0 + n;
  float sc = g3[col] * rsqrtf(v3[col] + EPSL);
  float sh = b3[col] - m3[col] * sc;
  float bv = cb[col];
#pragma unroll
  for (int r = 0; r < 8; ++r) {
    int row = row0 + rbase + r;
    int b_ = row / PP, p = row % PP;
    long idx = (long)b_ * CHW + (long)col * PP + p;
    out[idx] = x2[idx] + (acc[r] + bv) * sc + sh;
  }
}

// ======================= attention kernel =======================
// one wave per (b, h, 16-query block). 14 WMMAs for S, 14 for PV.
__global__ __launch_bounds__(32) void k_attn(const _Float16* __restrict__ Q,   // [bh,784,32]
                                             const _Float16* __restrict__ Kt,  // [bh,224,32]
                                             const _Float16* __restrict__ Vt,  // [bh,32,224]
                                             const float* __restrict__ pos_b,  // [8,784,196]
                                             _Float16* __restrict__ outp) {    // [b,784,256]
  __shared__ float    S[16 * NKP];
  __shared__ _Float16 Pm[16 * NKP];
  int lane = threadIdx.x & 31;
  int gid = blockIdx.x;
  int qb = gid % 49, bh = gid / 49;
  int h = bh % NH, b = bh / NH;
  int q0 = qb * 16;

  v16h aQ = load_A16x32(Q + ((long)bh * PP + q0) * 32, 32, lane);  // DK=32: one frag

  int n = lane & 15, rbase = (lane >> 4) << 3;
#pragma unroll
  for (int j = 0; j < NKP / 16; ++j) {
    v16h bm = load_B32x16(Kt + ((long)bh * NKP + j * 16) * 32, 32, lane);
    v8f c = {};
    c = wmma_f16(aQ, bm, c);
    int key = j * 16 + n;
#pragma unroll
    for (int r = 0; r < 8; ++r) {
      int m = rbase + r;
      float s = (key < NKV)
                  ? c[r] * SCALE + pos_b[((long)h * PP + q0 + m) * NKV + key]
                  : -1e30f;
      S[m * NKP + key] = s;
    }
  }
  __syncthreads();
  if (lane < 16) {
    float mx = -1e30f;
    for (int j = 0; j < NKP; ++j) mx = fmaxf(mx, S[lane * NKP + j]);
    float sum = 0.f;
    for (int j = 0; j < NKP; ++j) { float e = __expf(S[lane * NKP + j] - mx); S[lane * NKP + j] = e; sum += e; }
    float inv = 1.0f / sum;
    for (int j = 0; j < NKP; ++j) Pm[lane * NKP + j] = (_Float16)(S[lane * NKP + j] * inv);
  }
  __syncthreads();
#pragma unroll
  for (int t = 0; t < 2; ++t) {
    v8f acc = {};
#pragma unroll
    for (int kk = 0; kk < NKP; kk += 32) {
      v16h a = load_A16x32((const _Float16*)Pm + kk, NKP, lane);
      v16h bm = load_B32x16(Vt + ((long)bh * 32 + t * 16) * NKP + kk, NKP, lane);
      acc = wmma_f16(a, bm, acc);
    }
#pragma unroll
    for (int r = 0; r < 8; ++r) {
      int m = rbase + r;
      outp[((long)b * PP + q0 + m) * CC + h * 32 + t * 16 + n] = (_Float16)acc[r];
    }
  }
}

// ======================= host launch =======================
extern "C" void kernel_launch(void* const* d_in, const int* in_sizes, int n_in,
                              void* d_out, int out_size, void* d_ws, size_t ws_size,
                              hipStream_t stream) {
  const float* x     = (const float*)d_in[0];
  const float* lpu_w = (const float*)d_in[1];  const float* lpu_b = (const float*)d_in[2];
  const float* dw_w  = (const float*)d_in[3];  const float* dw_b  = (const float*)d_in[4];
  const float* wq    = (const float*)d_in[5];  const float* bq    = (const float*)d_in[6];
  const float* wk    = (const float*)d_in[7];  const float* bk    = (const float*)d_in[8];
  const float* wv    = (const float*)d_in[9];  const float* bv    = (const float*)d_in[10];
  const float* wo    = (const float*)d_in[11]; const float* bo    = (const float*)d_in[12];
  const float* pos_b = (const float*)d_in[13];
  const float* c1_w  = (const float*)d_in[14]; const float* c1_b  = (const float*)d_in[15];
  const float* bn1_g = (const float*)d_in[16]; const float* bn1_b = (const float*)d_in[17];
  const float* bn1_m = (const float*)d_in[18]; const float* bn1_v = (const float*)d_in[19];
  const float* dw2_w = (const float*)d_in[20]; const float* dw2_b = (const float*)d_in[21];
  const float* bn2_g = (const float*)d_in[22]; const float* bn2_b = (const float*)d_in[23];
  const float* bn2_m = (const float*)d_in[24]; const float* bn2_v = (const float*)d_in[25];
  const float* c2_w  = (const float*)d_in[26]; const float* c2_b  = (const float*)d_in[27];
  const float* bn3_g = (const float*)d_in[28]; const float* bn3_b = (const float*)d_in[29];
  const float* bn3_m = (const float*)d_in[30]; const float* bn3_v = (const float*)d_in[31];
  float* out = (float*)d_out;

  // ---- workspace bump layout ----
  char* base = (char*)d_ws;
  size_t off = 0;
  auto alloc = [&](size_t bytes) { void* p = base + off; off = (off + bytes + 255) & ~(size_t)255; return p; };
  float*    x1     = (float*)   alloc((size_t)BB * CHW * 4);
  float*    x2     = (float*)   alloc((size_t)BB * CHW * 4);
  float*    stats1 = (float*)   alloc(BB * 2 * 4);
  float*    stats2 = (float*)   alloc(BB * 2 * 4);
  _Float16* wq16   = (_Float16*)alloc((size_t)CC * CC * 2);
  _Float16* wk16   = (_Float16*)alloc((size_t)CC * CC * 2);
  _Float16* wv16   = (_Float16*)alloc((size_t)CC * CC * 2);
  _Float16* wo16   = (_Float16*)alloc((size_t)CC * CC * 2);
  _Float16* c116   = (_Float16*)alloc((size_t)CM * CC * 2);
  _Float16* c216   = (_Float16*)alloc((size_t)CC * CM * 2);
  _Float16* xn16   = (_Float16*)alloc((size_t)BB * PP * CC * 2);
  _Float16* xn216  = (_Float16*)alloc((size_t)BB * PP * CC * 2);
  _Float16* kvr16  = (_Float16*)alloc((size_t)BB * NKP * CC * 2);
  _Float16* Q16    = (_Float16*)alloc((size_t)BB * NH * PP * 32 * 2);
  _Float16* K16    = (_Float16*)alloc((size_t)BB * NH * NKP * 32 * 2);
  _Float16* Vt16   = (_Float16*)alloc((size_t)BB * NH * 32 * NKP * 2);
  _Float16* att16  = (_Float16*)alloc((size_t)BB * PP * CC * 2);
  _Float16* t1     = (_Float16*)alloc((size_t)BB * CM * PP * 2);
  _Float16* t2     = (_Float16*)alloc((size_t)BB * PP * CM * 2);

  const int M  = BB * PP;    // 25088 token rows
  const int Mk = BB * NKP;   // 7168 padded key rows

  // weights -> fp16
  k_cvt16<<<(CC * CC + 255) / 256, 256, 0, stream>>>(wq, wq16, CC * CC);
  k_cvt16<<<(CC * CC + 255) / 256, 256, 0, stream>>>(wk, wk16, CC * CC);
  k_cvt16<<<(CC * CC + 255) / 256, 256, 0, stream>>>(wv, wv16, CC * CC);
  k_cvt16<<<(CC * CC + 255) / 256, 256, 0, stream>>>(wo, wo16, CC * CC);
  k_cvt16<<<(CM * CC + 255) / 256, 256, 0, stream>>>(c1_w, c116, CM * CC);
  k_cvt16<<<(CC * CM + 255) / 256, 256, 0, stream>>>(c2_w, c216, CC * CM);

  // LPU + LN1 + packs
  k_lpu<<<(BB * CC * PP + 255) / 256, 256, 0, stream>>>(x, lpu_w, lpu_b, x1);
  k_ln_stats<<<BB, 256, 0, stream>>>(x1, stats1);
  k_ln_pack<<<(int)(((long)BB * CHW + 255) / 256), 256, 0, stream>>>(x1, stats1, xn16);
  k_kvconv<<<(BB * NKP * CC + 255) / 256, 256, 0, stream>>>(x1, dw_w, dw_b, kvr16);

  // Q/K/V projections (WMMA)
  k_gemm_proj<<<((M >> 4) * (CC >> 4) + 7) / 8, 256, 0, stream>>>(xn16, wq16, bq, Q16, M, CC, PP, 0);
  k_gemm_proj<<<((Mk >> 4) * (CC >> 4) + 7) / 8, 256, 0, stream>>>(kvr16, wk16, bk, K16, Mk, CC, NKP, 0);
  k_gemm_proj<<<((Mk >> 4) * (CC >> 4) + 7) / 8, 256, 0, stream>>>(kvr16, wv16, bv, Vt16, Mk, CC, NKP, 1);

  // attention (WMMA S and PV, fused softmax in LDS)
  k_attn<<<BB * NH * (PP / 16), 32, 0, stream>>>(Q16, K16, Vt16, pos_b, att16);

  // output projection + residual (WMMA)
  k_gemm_oproj<<<((M >> 4) * (CC >> 4) + 7) / 8, 256, 0, stream>>>(att16, wo16, bo, x1, x2, M);

  // LN2 + IRFFN
  k_ln_stats<<<BB, 256, 0, stream>>>(x2, stats2);
  k_ln_pack<<<(int)(((long)BB * CHW + 255) / 256), 256, 0, stream>>>(x2, stats2, xn216);
  k_gemm_ffn1<<<((M >> 4) * (CM >> 4) + 7) / 8, 256, 0, stream>>>(xn216, c116, c1_b,
                                                                  bn1_g, bn1_b, bn1_m, bn1_v, t1, M);
  k_dw2<<<(int)(((long)BB * CM * PP + 255) / 256), 256, 0, stream>>>(t1, dw2_w, dw2_b,
                                                                     bn2_g, bn2_b, bn2_m, bn2_v, t2);
  k_gemm_ffn2<<<((M >> 4) * (CC >> 4) + 7) / 8, 256, 0, stream>>>(t2, c216, c2_b,
                                                                  bn3_g, bn3_b, bn3_m, bn3_v, x2, out, M);
}
